// TrendLSTM_5394478923827
// MI455X (gfx1250) — matrix operand order repaired and checked by reading, not compiled, not run on hardware
//
#include <hip/hip_runtime.h>

typedef __attribute__((ext_vector_type(16))) _Float16 v16h;
typedef __attribute__((ext_vector_type(8)))  _Float16 v8h;
typedef __attribute__((ext_vector_type(8)))  float    v8f;

#define HIDDEN 64
#define MTILE  16
#define XCHUNK 64

// Hardware reciprocal (v_rcp_f32, ~1 ulp) instead of the IEEE division
// expansion (v_div_scale/v_div_fmas/v_div_fixup ~10 VALU ops).
__device__ __forceinline__ float fast_rcp(float x) {
    return __builtin_amdgcn_rcpf(x);
}
__device__ __forceinline__ float fast_sigmoid(float x) {
    return fast_rcp(1.0f + __expf(-x));          // v_exp_f32 + v_rcp_f32
}
__device__ __forceinline__ float fast_tanh(float x) {
    return 1.0f - 2.0f * fast_rcp(1.0f + __expf(2.0f * x));
}

// One workgroup = 4 waves = 16 batch rows, resident across all T steps.
// Wave w computes gate w (i/f/g/o): gates[16, 64] = h[16,64] @ W_hh[64w..64w+64, :]^T
// via 4 N-tiles x 2 K-halves of v_wmma_f32_16x16x32_f16, W_hh pre-swizzled into
// register-resident B fragments. h crosses steps through LDS (b128-friendly layout).
__global__ __launch_bounds__(128, 1) void lstm_wmma_kernel(
    const float* __restrict__ x,    const float* __restrict__ W_ih,
    const float* __restrict__ W_hh, const float* __restrict__ b_ih,
    const float* __restrict__ b_hh, const float* __restrict__ W_fc,
    const float* __restrict__ b_fc, float* __restrict__ out, int T)
{
    __shared__ __align__(16) float    gatebuf[MTILE * 4 * HIDDEN]; // 16x256 f32 = 16KB
    __shared__ __align__(16) _Float16 hbuf[MTILE * HIDDEN];        // 16x64 f16 = 2KB
    __shared__ __align__(16) float    xchunk[MTILE * XCHUNK];      // 16x64 f32 = 4KB
    __shared__ float red[128];

    const int tid     = threadIdx.x;
    const int wave    = tid >> 5;       // gate id: 0=i 1=f 2=g 3=o
    const int lane    = tid & 31;
    const int baserow = blockIdx.x * MTILE;

    // ---- preload W_hh^T slices as B fragments (resident all T steps) ----
    // B 32x16 f16 fragment, contiguous k order used here: lanes 0-15 hold K=0..15 (col = lane), lanes 16-31 hold
    // K=16..31; VGPR v packs K=2v,2v+1 -> element e of v16h is K = kb + e.
    const int n  = lane & 15;
    const int kb = (lane >> 4) * 16;
    v16h bfrag[4][2];
    float wih_c[4], bias_c[4];
    #pragma unroll
    for (int nt = 0; nt < 4; ++nt) {
        const int col = wave * 64 + nt * 16 + n;
        const float* wr = W_hh + col * HIDDEN;     // row of W_hh = column of B
        #pragma unroll
        for (int kh = 0; kh < 2; ++kh)
            #pragma unroll
            for (int e = 0; e < 16; ++e)
                bfrag[nt][kh][e] = (_Float16)wr[kh * 32 + kb + e];
        wih_c[nt]  = W_ih[col];
        bias_c[nt] = b_ih[col] + b_hh[col];
    }

    // element ownership for the cross-gate update: 16x64 elems / 128 threads = 8 each,
    // all 8 in one row: row = tid/8, hid = (tid%8)*8 + j
    const int erow = tid >> 3;
    const int ehid = (tid & 7) * 8;
    float c_reg[8], h_reg[8];
    #pragma unroll
    for (int j = 0; j < 8; ++j) { c_reg[j] = 0.f; h_reg[j] = 0.f; }

    for (int i = tid; i < MTILE * HIDDEN; i += 128) hbuf[i] = (_Float16)0.f;
    __syncthreads();

    const int m   = lane & 15;           // A-matrix row (batch row in tile)
    const int sel = (lane >> 4) << 3;    // lane half selects K sub-block

    for (int t = 0; t < T; ++t) {
        const int tt = t & (XCHUNK - 1);
        if (tt == 0) {                   // stage 64 timesteps of x, coalesced
            for (int i = tid; i < MTILE * XCHUNK; i += 128) {
                const int r = i >> 6, q = i & 63, tg = t + q;
                xchunk[i] = (tg < T) ? x[(size_t)(baserow + r) * T + tg] : 0.f;
            }
            __syncthreads();
        }

        // ---- A fragments from h: contiguous k order, same as B ----
        // lane<16: elems 0..15 = K0..15 ; lane>=16: K16..31 (matches B)
        const _Float16* hr = &hbuf[m * HIDDEN];
        const v8h p0 = *(const v8h*)(hr +      2 * sel);
        const v8h p1 = *(const v8h*)(hr + 8  + 2 * sel);
        const v8h p2 = *(const v8h*)(hr + 32 + 2 * sel);
        const v8h p3 = *(const v8h*)(hr + 40 + 2 * sel);
        v16h a0, a1;
        #pragma unroll
        for (int i = 0; i < 8; ++i) {
            a0[i] = p0[i]; a0[8 + i] = p1[i];
            a1[i] = p2[i]; a1[8 + i] = p3[i];
        }

        // per-row x for accumulator seeding (C layout: VGPR r -> row r + 8*(lane>=16))
        const int rowbase = (lane >> 4) << 3;
        float xv[8];
        #pragma unroll
        for (int r = 0; r < 8; ++r)
            xv[r] = xchunk[(rowbase + r) * XCHUNK + tt];

        #pragma unroll
        for (int nt = 0; nt < 4; ++nt) {
            v8f acc;
            #pragma unroll
            for (int r = 0; r < 8; ++r)
                acc[r] = xv[r] * wih_c[nt] + bias_c[nt];   // x*W_ih + b_ih + b_hh
            acc = __builtin_amdgcn_wmma_f32_16x16x32_f16(
                      false, a0, false, bfrag[nt][0], (short)0, acc, false, false);
            acc = __builtin_amdgcn_wmma_f32_16x16x32_f16(
                      false, a1, false, bfrag[nt][1], (short)0, acc, false, false);
            const int colg = wave * 64 + nt * 16 + n;
            #pragma unroll
            for (int r = 0; r < 8; ++r)
                gatebuf[(rowbase + r) * (4 * HIDDEN) + colg] = acc[r];
        }
        __syncthreads();

        // ---- cross-gate elementwise update (c,h in f32 registers) ----
        const float* grow = &gatebuf[erow * (4 * HIDDEN)];
        #pragma unroll
        for (int j = 0; j < 8; ++j) {
            const float iv = fast_sigmoid(grow[             ehid + j]);
            const float fv = fast_sigmoid(grow[    HIDDEN + ehid + j]);
            const float gv = fast_tanh   (grow[2 * HIDDEN + ehid + j]);
            const float ov = fast_sigmoid(grow[3 * HIDDEN + ehid + j]);
            c_reg[j] = fv * c_reg[j] + iv * gv;
            h_reg[j] = ov * fast_tanh(c_reg[j]);
            hbuf[erow * HIDDEN + ehid + j] = (_Float16)h_reg[j];
        }
        __syncthreads();
    }

    // ---- out[b] = h_T[b,:] . W_fc + b_fc ----
    float partial = 0.f;
    #pragma unroll
    for (int j = 0; j < 8; ++j)
        partial += h_reg[j] * W_fc[ehid + j];
    red[tid] = partial;
    __syncthreads();
    if (tid < MTILE) {
        float s = b_fc[0];
        #pragma unroll
        for (int k = 0; k < 8; ++k) s += red[tid * 8 + k];
        out[baserow + tid] = s;
    }
}

extern "C" void kernel_launch(void* const* d_in, const int* in_sizes, int n_in,
                              void* d_out, int out_size, void* d_ws, size_t ws_size,
                              hipStream_t stream) {
    const float* x    = (const float*)d_in[0];
    const float* W_ih = (const float*)d_in[1];
    const float* W_hh = (const float*)d_in[2];
    const float* b_ih = (const float*)d_in[3];
    const float* b_hh = (const float*)d_in[4];
    const float* W_fc = (const float*)d_in[5];
    const float* b_fc = (const float*)d_in[6];
    float* out = (float*)d_out;

    const int B = out_size;            // [B,1] output
    const int T = in_sizes[0] / B;     // x is [B,T,1]

    dim3 grid(B / MTILE), block(128);
    lstm_wmma_kernel<<<grid, block, 0, stream>>>(x, W_ih, W_hh, b_ih, b_hh,
                                                 W_fc, b_fc, out, T);
}
